// PIGNet_30674656428268
// MI455X (gfx1250) — matrix operand order repaired and checked by reading, not compiled, not run on hardware
//
#include <hip/hip_runtime.h>
#include <hip/hip_bf16.h>
#include <math.h>

#define N_NODES  50000
#define N_INTRA  800000
#define N_INTER  400000
#define N_IEDGE  600000
#define N_GRAPHS 128
#define F_IN     54
#define DIM      128
#define DEV_VDW_COEFF 0.2f
#define EPS_LO   0.0178f
#define EPS_HI   0.2f
#define RANGE_LO 0.5f
#define RANGE_HI 5.0f

typedef __attribute__((ext_vector_type(16))) _Float16 v16h;
typedef __attribute__((ext_vector_type(8)))  _Float16 v8h;
typedef __attribute__((ext_vector_type(8)))  float    v8f;

// ---------------- WMMA fragment loaders (CDNA5 16-bit layouts) ----------------
// A 16x32 (MxK): lane = M row (lane&15), halves 0..7 -> K = k0+0..7,
// halves 8..15 -> K = k0+16..23, k0 = 8 for lanes >= 16.
__device__ __forceinline__ v16h load_a_frag(const _Float16* row, int kbase, int lane) {
    int k0 = kbase + ((lane & 16) ? 8 : 0);
    union { v16h v; v8h h[2]; } u;
    u.h[0] = *(const v8h*)(row + k0);
    u.h[1] = *(const v8h*)(row + k0 + 16);
    return u.v;
}
// B 32x16 (KxN) from transposed (out-major) weights: lane = N column (lane&15),
// halves 0..7 -> K = ks+0..7, halves 8..15 -> K = ks+16..23, ks = kt32 (+8 for lanes >= 16).
__device__ __forceinline__ v16h load_b_frag(const _Float16* wrow, int kt32, int lane) {
    int ks = kt32 + ((lane & 16) ? 8 : 0);
    union { v16h v; v8h h[2]; } u;
    u.h[0] = *(const v8h*)(wrow + ks);
    u.h[1] = *(const v8h*)(wrow + ks + 16);
    return u.v;
}

// ordered-uint encoding for float atomicMax
__device__ __forceinline__ unsigned f2ord(float f) {
    unsigned b = __float_as_uint(f);
    return (b & 0x80000000u) ? ~b : (b | 0x80000000u);
}
__device__ __forceinline__ float ord2f(unsigned k) {
    unsigned b = (k & 0x80000000u) ? (k & 0x7FFFFFFFu) : ~k;
    return __uint_as_float(b);
}

// ---------------- utility kernels ----------------
__global__ void pignet_fill_u32(unsigned* p, unsigned v, int n) {
    int t = blockIdx.x * blockDim.x + threadIdx.x;
    if (t < n) p[t] = v;
}
__global__ void pignet_f2h(const float* s, _Float16* d, int n) {
    int t = blockIdx.x * blockDim.x + threadIdx.x;
    if (t < n) d[t] = (_Float16)s[t];
}
// W [K][N] f32 row-major  ->  Wt [N][K] f16
__global__ void pignet_wconv(const float* W, _Float16* Wt, int K, int N) {
    int t = blockIdx.x * blockDim.x + threadIdx.x;
    if (t >= K * N) return;
    int k = t / N, n = t % N;
    Wt[(size_t)n * K + k] = (_Float16)W[t];
}
__global__ void pignet_embed(const float* x, const float* W, float* h) {
    int t = blockIdx.x * blockDim.x + threadIdx.x;  // N_NODES*DIM exact
    int n = t >> 7, d = t & 127;
    const float* xr = x + (size_t)n * F_IN;
    float a = 0.f;
    #pragma unroll
    for (int k = 0; k < F_IN; ++k) a += xr[k] * W[k * DIM + d];
    h[t] = a;
}

// ---------------- WMMA GEMM: out[M,128] = f16(A) @ W(^T stored) + bias ----------------
// block = 256 (8 waves); block b handles rows [16b,16b+16); wave w -> cols [16w,16w+16)
// Each wave reads a disjoint 16-row slice of Wt, so no LDS staging pays off here.
__global__ void pignet_gemm128(const _Float16* __restrict__ A16,
                               const _Float16* __restrict__ Wt,   // [128][128] f16, out-major
                               const float* __restrict__ bias,
                               float* __restrict__ out, _Float16* __restrict__ out16,
                               int relu) {
    int lane = threadIdx.x & 31;
    int wv   = threadIdx.x >> 5;
    int m0   = blockIdx.x * 16;
    int n0   = wv * 16;
    const _Float16* arow = A16 + (size_t)(m0 + (lane & 15)) * DIM;
    const _Float16* brow = Wt  + (size_t)(n0 + (lane & 15)) * DIM;
    v8f acc = {0.f,0.f,0.f,0.f,0.f,0.f,0.f,0.f};
    #pragma unroll
    for (int kt = 0; kt < 4; ++kt) {
        v16h a = load_a_frag(arow, kt * 32, lane);
        v16h b = load_b_frag(brow, kt * 32, lane);
        acc = __builtin_amdgcn_wmma_f32_16x16x32_f16(false, a, false, b,
                                                     (short)0, acc, false, false);
    }
    int n = n0 + (lane & 15);
    float bz = bias ? bias[n] : 0.f;
    int mh = (lane >> 4) * 8;   // D layout: vgpr v, lane -> M = v + 8*(lane/16), N = lane&15
    #pragma unroll
    for (int v = 0; v < 8; ++v) {
        float xv = acc[v] + bz;
        if (relu) xv = fmaxf(xv, 0.f);
        size_t o = (size_t)(m0 + mh + v) * DIM + n;
        out[o] = xv;
        if (out16) out16[o] = (_Float16)xv;
    }
}

// ---------------- GAT softmax pieces ----------------
__global__ void pignet_logits(const float* __restrict__ hA, const float* __restrict__ h1,
                              const int* __restrict__ src, const int* __restrict__ dst,
                              float* __restrict__ logits, unsigned* __restrict__ mxu, int E) {
    int e = blockIdx.x * blockDim.x + threadIdx.x;
    if (e >= E) return;
    int s = src[e], d = dst[e];
    const float4* pa = (const float4*)(hA + (size_t)d * DIM);
    const float4* pb = (const float4*)(h1 + (size_t)s * DIM);
    float acc = 0.f;
    #pragma unroll 8
    for (int k = 0; k < 32; ++k) {
        float4 a = pa[k], b = pb[k];
        acc += a.x*b.x + a.y*b.y + a.z*b.z + a.w*b.w;
    }
    logits[e] = acc;
    atomicMax(mxu + d, f2ord(acc));
}
__global__ void pignet_ex(float* __restrict__ exv, const int* __restrict__ dst,
                          const unsigned* __restrict__ mxu, float* __restrict__ den, int E) {
    int e = blockIdx.x * blockDim.x + threadIdx.x;
    if (e >= E) return;
    int d = dst[e];
    float m = ord2f(mxu[d]);
    if ((__float_as_uint(m) & 0x7F800000u) == 0x7F800000u) m = 0.f;  // non-finite -> 0
    float v = expf(exv[e] - m);
    exv[e] = v;
    atomicAdd(den + d, v);
}
// hp[dst] += alpha * h1[src]   (alpha = 1 when exv == nullptr)
__global__ void pignet_scatter(const float* __restrict__ exv, const float* __restrict__ den,
                               const float* __restrict__ h1,
                               const int* __restrict__ src, const int* __restrict__ dst,
                               float* __restrict__ hp, int E) {
    int gid = blockIdx.x * blockDim.x + threadIdx.x;
    int e = gid >> 5;
    if (e >= E) return;
    int fg = (gid & 31) * 4;
    int s = src[e], d = dst[e];
    float alpha = exv ? (exv[e] / (den[d] + 1e-16f)) : 1.f;
    float4 v = *(const float4*)(h1 + (size_t)s * DIM + fg);
    float* o = hp + (size_t)d * DIM + fg;
    atomicAdd(o + 0, alpha * v.x);
    atomicAdd(o + 1, alpha * v.y);
    atomicAdd(o + 2, alpha * v.z);
    atomicAdd(o + 3, alpha * v.w);
}
// h = z*h + (1-z)*msg', msg' = relu(msg) if relu_msg; z = sigmoid([h,msg']@gw + gb)
__global__ void pignet_blend(float* __restrict__ h, const float* __restrict__ msg,
                             const float* __restrict__ gw, const float* __restrict__ gb,
                             int relu_msg, int Nn) {
    int lane = threadIdx.x & 31;
    int node = blockIdx.x * (blockDim.x >> 5) + (threadIdx.x >> 5);
    if (node >= Nn) return;
    float* hx = h + (size_t)node * DIM;
    const float* mr = msg + (size_t)node * DIM;
    float dotv = 0.f;
    #pragma unroll
    for (int t = 0; t < 4; ++t) {
        int f = lane + t * 32;
        float m = mr[f]; if (relu_msg) m = fmaxf(m, 0.f);
        dotv += hx[f] * gw[f] + m * gw[DIM + f];
    }
    #pragma unroll
    for (int m = 1; m < 32; m <<= 1) dotv += __shfl_xor(dotv, m, 32);
    float z = 1.f / (1.f + expf(-(dotv + gb[0])));
    #pragma unroll
    for (int t = 0; t < 4; ++t) {
        int f = lane + t * 32;
        float m = mr[f]; if (relu_msg) m = fmaxf(m, 0.f);
        hx[f] = z * hx[f] + (1.f - z) * m;
    }
}

// ---------------- per-i-edge geometry ----------------
__global__ void pignet_dist(const float* __restrict__ pos, const int* __restrict__ ei,
                            const int* __restrict__ ej, float* __restrict__ Darr, int E) {
    int e = blockIdx.x * blockDim.x + threadIdx.x;
    if (e >= E) return;
    int i = ei[e], j = ej[e];
    float dx = pos[3*i+0] - pos[3*j+0];
    float dy = pos[3*i+1] - pos[3*j+1];
    float dz = pos[3*i+2] - pos[3*j+2];
    Darr[e] = sqrtf(dx*dx + dy*dy + dz*dz + 1e-10f);
}

// ---------------- per-edge MLP: [h[i],h[j]](256) -> relu -> 128 -> 1 ----------------
// wave per 16-edge tile; hidden [16,128] held in 8 x v8f WMMA accumulators.
// W1t (64 KB) is staged once per block into LDS: 8 waves would otherwise each
// stream the full matrix (512 KB/block of redundant vmem traffic).
// mode 0: eps = sigmoid(s)*(HI-LO)+LO ; mode 1: dvdw = tanh(s)*0.2*rmask(D)
__global__ void pignet_edge_mlp(const _Float16* __restrict__ hh,
                                const int* __restrict__ ei, const int* __restrict__ ej,
                                const _Float16* __restrict__ W1t,   // [128][256] f16
                                const float* __restrict__ b1, const float* __restrict__ w2,
                                const float* __restrict__ b2, const float* __restrict__ Darr,
                                float* __restrict__ out, int mode, int ntiles) {
    __shared__ _Float16 sW[DIM * 2 * DIM];   // 128 rows x 256 cols, 64 KB
    {
        const uint4* g = (const uint4*)W1t;
        uint4* s = (uint4*)sW;
        for (int t = threadIdx.x; t < (DIM * 2 * DIM) / 8; t += blockDim.x)
            s[t] = g[t];
    }
    __syncthreads();

    int lane = threadIdx.x & 31;
    int tile = blockIdx.x * (blockDim.x >> 5) + (threadIdx.x >> 5);
    if (tile < ntiles) {                      // wave-uniform: EXEC all-ones for WMMA
        int r  = lane & 15;
        int er = tile * 16 + r;
        int ni = ei[er], nj = ej[er];
        __builtin_prefetch(hh + (size_t)nj * DIM, 0, 1);  // global_prefetch for j-half of K loop
        v8f zero = {0.f,0.f,0.f,0.f,0.f,0.f,0.f,0.f};
        v8f acc[8];
        #pragma unroll
        for (int t = 0; t < 8; ++t) acc[t] = zero;
        #pragma unroll
        for (int kt = 0; kt < 8; ++kt) {      // K = 256 in steps of 32; first 128 = h[i], rest = h[j]
            const _Float16* row = hh + (size_t)((kt < 4) ? ni : nj) * DIM;
            v16h a = load_a_frag(row, (kt & 3) * 32, lane);
            #pragma unroll
            for (int t = 0; t < 8; ++t) {
                const _Float16* brow = sW + (size_t)(t * 16 + r) * 256;
                v16h b = load_b_frag(brow, kt * 32, lane);
                acc[t] = __builtin_amdgcn_wmma_f32_16x16x32_f16(false, a, false, b,
                                                                (short)0, acc[t], false, false);
            }
        }
        // second layer: relu(hidden + b1) . w2, reduced over the 16 lanes of each half-wave
        float part[8];
        #pragma unroll
        for (int v = 0; v < 8; ++v) part[v] = 0.f;
        #pragma unroll
        for (int t = 0; t < 8; ++t) {
            int f = t * 16 + r;
            float b1f = b1[f], w2f = w2[f];
            #pragma unroll
            for (int v = 0; v < 8; ++v)
                part[v] += fmaxf(acc[t][v] + b1f, 0.f) * w2f;
        }
        #pragma unroll
        for (int m = 1; m < 16; m <<= 1) {
            #pragma unroll
            for (int v = 0; v < 8; ++v) part[v] += __shfl_xor(part[v], m, 32);
        }
        if (r == 0) {                          // lanes 0 (edges 0..7) and 16 (edges 8..15)
            int mh = (lane >> 4) * 8;
            #pragma unroll
            for (int v = 0; v < 8; ++v) {
                int e = tile * 16 + mh + v;
                float s = part[v] + b2[0];
                float o;
                if (mode == 0) {
                    o = (1.f / (1.f + expf(-s))) * (EPS_HI - EPS_LO) + EPS_LO;
                } else {
                    float D  = Darr[e];
                    float rm = (D >= RANGE_LO && D <= RANGE_HI) ? 1.f : 0.f;
                    o = tanhf(s) * DEV_VDW_COEFF * rm;
                }
                out[e] = o;
            }
        }
    }
}

// ---------------- energy terms + segment-sum over graphs ----------------
__global__ void pignet_energy(const float* __restrict__ Darr, const float* __restrict__ vdw,
                              const float* __restrict__ epsArr, const float* __restrict__ dvdwArr,
                              const int* __restrict__ ei, const int* __restrict__ ej,
                              const int* __restrict__ batch,
                              const unsigned char* __restrict__ mM, const unsigned char* __restrict__ mHD,
                              const unsigned char* __restrict__ mHAc, const unsigned char* __restrict__ mHY,
                              const float* __restrict__ hbC, const float* __restrict__ hyC,
                              float* __restrict__ energ, int E) {
    __shared__ float sE[N_GRAPHS * 4];
    for (int t = threadIdx.x; t < N_GRAPHS * 4; t += blockDim.x) sE[t] = 0.f;
    __syncthreads();
    int e = blockIdx.x * blockDim.x + threadIdx.x;
    if (e < E) {
        int i = ei[e], j = ej[e];
        float D  = Darr[e];
        float rm = (D >= RANGE_LO && D <= RANGE_HI) ? 1.f : 0.f;
        float R   = vdw[i] + vdw[j] + dvdwArr[e];
        float eps = epsArr[e];
        float dr  = fmaxf(D / R, 0.5f);
        float nn  = (dr < 1.f) ? 10.f : 6.f;
        float lj  = eps * (powf(dr, -2.f * nn) - 2.f * powf(dr, -nn));
        float hb = hbC[0], hy = hyC[0];
        float minHB = -hb * hb, minHP = -hy * hy;
        float dmr = D - R;
        float f1 = fminf(fmaxf(dmr / (-0.7f), 0.f), 1.f);          // HBOND/METAL cuts (-0.7, 0.0)
        float f3 = fminf(fmaxf((dmr - 1.5f) / (-1.0f), 0.f), 1.f); // HYDRO cut (0.5, 1.5)
        bool bm_i = mM[i] != 0, bm_j = mM[j] != 0;
        bool noM  = !(bm_i || bm_j);
        bool hd_i = mHD[i] != 0, hd_j = mHD[j] != 0;
        bool ha_i = mHAc[i] != 0, ha_j = mHAc[j] != 0;
        bool mskHB = ((hd_i && ha_j) || (ha_i && hd_j)) && noM;
        bool mskMT = (bm_i && ha_j) || (ha_i && bm_j);
        bool mskHY = (mHY[i] != 0) && (mHY[j] != 0) && noM;
        int g = batch[i];
        atomicAdd(&sE[g*4+0], lj          * (noM   ? rm : 0.f));
        atomicAdd(&sE[g*4+1], minHB * f1  * (mskHB ? rm : 0.f));
        atomicAdd(&sE[g*4+2], minHB * f1  * (mskMT ? rm : 0.f));
        atomicAdd(&sE[g*4+3], minHP * f3  * (mskHY ? rm : 0.f));
    }
    __syncthreads();
    for (int t = threadIdx.x; t < N_GRAPHS * 4; t += blockDim.x) {
        float v = sE[t];
        if (v != 0.f) atomicAdd(&energ[t], v);
    }
}
__global__ void pignet_final(const float* __restrict__ energ, const float* __restrict__ rotor,
                             const float* __restrict__ roC, float* __restrict__ out) {
    int t = blockIdx.x * blockDim.x + threadIdx.x;
    if (t < N_GRAPHS * 4) {
        int g = t >> 2;
        float pen = 1.f + roC[0] * roC[0] * rotor[g];
        out[t] = energ[t] / pen;
    }
}

// ---------------- driver ----------------
extern "C" void kernel_launch(void* const* d_in, const int* in_sizes, int n_in,
                              void* d_out, int out_size, void* d_ws, size_t ws_size,
                              hipStream_t stream) {
    (void)in_sizes; (void)n_in; (void)out_size; (void)ws_size;
    const float* x      = (const float*)d_in[0];
    const float* pos    = (const float*)d_in[1];
    const float* vdw    = (const float*)d_in[2];
    const float* rotor  = (const float*)d_in[3];
    const int*   eIntra = (const int*)d_in[4];
    const int*   eInter = (const int*)d_in[5];
    const int*   eIe    = (const int*)d_in[6];
    const int*   batch  = (const int*)d_in[7];
    const unsigned char* mMetal = (const unsigned char*)d_in[8];
    const unsigned char* mHD    = (const unsigned char*)d_in[9];
    const unsigned char* mHAc   = (const unsigned char*)d_in[10];
    const unsigned char* mHY    = (const unsigned char*)d_in[11];
    const float* embedW = (const float*)d_in[12];
    const float* gatW   = (const float*)d_in[13];
    const float* gatWb  = (const float*)d_in[14];
    const float* gatA   = (const float*)d_in[15];
    const float* gatGw  = (const float*)d_in[16];
    const float* gatGb  = (const float*)d_in[17];
    const float* intW   = (const float*)d_in[18];
    const float* intWb  = (const float*)d_in[19];
    const float* intGw  = (const float*)d_in[20];
    const float* intGb  = (const float*)d_in[21];
    const float* epsW1  = (const float*)d_in[22];
    const float* epsB1  = (const float*)d_in[23];
    const float* epsW2  = (const float*)d_in[24];
    const float* epsB2  = (const float*)d_in[25];
    const float* dvW1   = (const float*)d_in[26];
    const float* dvB1   = (const float*)d_in[27];
    const float* dvW2   = (const float*)d_in[28];
    const float* dvB2   = (const float*)d_in[29];
    const float* hbC    = (const float*)d_in[30];
    const float* hyC    = (const float*)d_in[31];
    const float* roC    = (const float*)d_in[32];

    float* out     = (float*)d_out;
    float* outE    = out;                 // [128,4]
    float* outDvdw = out + N_GRAPHS * 4;  // [600000]

    // workspace bump allocator (~109 MB)
    char* ws = (char*)d_ws;
    size_t cur = 0;
    auto alloc = [&](size_t b) -> void* {
        void* p = ws + cur;
        cur = (cur + b + 255) & ~(size_t)255;
        return p;
    };
    const size_t ND = (size_t)N_NODES * DIM;
    float*    H    = (float*)alloc(ND * 4);
    _Float16* HHf  = (_Float16*)alloc(ND * 2);
    float*    H1   = (float*)alloc(ND * 4);
    _Float16* H1H  = (_Float16*)alloc(ND * 2);
    float*    HA   = (float*)alloc(ND * 4);           // also h_prime / m (reused)
    float*    EX   = (float*)alloc((size_t)N_INTRA * 4);  // logits/ex; later eps per i-edge
    unsigned* MXU  = (unsigned*)alloc((size_t)N_NODES * 4);
    float*    DEN  = (float*)alloc((size_t)N_NODES * 4);
    float*    DA   = (float*)alloc((size_t)N_IEDGE * 4);
    _Float16* WgT  = (_Float16*)alloc((size_t)3 * DIM * DIM * 2);
    _Float16* AgT  = (_Float16*)alloc((size_t)3 * DIM * DIM * 2);
    _Float16* WiT  = (_Float16*)alloc((size_t)3 * DIM * DIM * 2);
    _Float16* We1T = (_Float16*)alloc((size_t)2 * DIM * DIM * 2);  // 128x256
    _Float16* Wd1T = (_Float16*)alloc((size_t)2 * DIM * DIM * 2);
    float*    EG   = (float*)alloc((size_t)N_GRAPHS * 4 * 4);

    const int T = 256;
    // weight conversions (f32 [K][N] -> f16 [N][K])
    for (int l = 0; l < 3; ++l) {
        pignet_wconv<<<(DIM*DIM+T-1)/T, T, 0, stream>>>(gatW + l*DIM*DIM, WgT + l*DIM*DIM, DIM, DIM);
        pignet_wconv<<<(DIM*DIM+T-1)/T, T, 0, stream>>>(gatA + l*DIM*DIM, AgT + l*DIM*DIM, DIM, DIM);
        pignet_wconv<<<(DIM*DIM+T-1)/T, T, 0, stream>>>(intW + l*DIM*DIM, WiT + l*DIM*DIM, DIM, DIM);
    }
    pignet_wconv<<<(2*DIM*DIM+T-1)/T, T, 0, stream>>>(epsW1, We1T, 2*DIM, DIM);
    pignet_wconv<<<(2*DIM*DIM+T-1)/T, T, 0, stream>>>(dvW1,  Wd1T, 2*DIM, DIM);

    // embedding
    pignet_embed<<<(int)(ND / T), T, 0, stream>>>(x, embedW, H);

    const int gF2H   = (int)(ND / T);                 // 25000
    const int gGemm  = N_NODES / 16;                  // 3125
    const int gNode  = (N_NODES + T - 1) / T;         // fills
    const int gIntra = N_INTRA / T;                   // 3125
    const int gScatI = (int)(((size_t)N_INTRA * 32) / T);
    const int gScatC = (int)(((size_t)N_INTER * 32) / T);
    const int gBlend = N_NODES / 8;                   // 6250

    // -------- 3 gated-GAT layers (intra edges) --------
    const int* srcA = eIntra;
    const int* dstA = eIntra + N_INTRA;
    for (int l = 0; l < 3; ++l) {
        pignet_f2h<<<gF2H, T, 0, stream>>>(H, HHf, (int)ND);
        pignet_gemm128<<<gGemm, T, 0, stream>>>(HHf, WgT + l*DIM*DIM, gatWb + l*DIM, H1, H1H, 0);
        pignet_gemm128<<<gGemm, T, 0, stream>>>(H1H, AgT + l*DIM*DIM, nullptr, HA, nullptr, 0);
        pignet_fill_u32<<<gNode, T, 0, stream>>>(MXU, 0u, N_NODES);
        pignet_fill_u32<<<gNode, T, 0, stream>>>((unsigned*)DEN, 0u, N_NODES);
        pignet_logits<<<gIntra, T, 0, stream>>>(HA, H1, srcA, dstA, EX, MXU, N_INTRA);
        pignet_ex<<<gIntra, T, 0, stream>>>(EX, dstA, MXU, DEN, N_INTRA);
        pignet_fill_u32<<<gF2H, T, 0, stream>>>((unsigned*)HA, 0u, (int)ND);
        pignet_scatter<<<gScatI, T, 0, stream>>>(EX, DEN, H1, srcA, dstA, HA, N_INTRA);
        pignet_blend<<<gBlend, T, 0, stream>>>(H, HA, gatGw + l*2*DIM, gatGb + l, 1, N_NODES);
    }

    // -------- 3 interaction layers (inter edges) --------
    const int* srcC = eInter;
    const int* dstC = eInter + N_INTER;
    for (int l = 0; l < 3; ++l) {
        pignet_f2h<<<gF2H, T, 0, stream>>>(H, HHf, (int)ND);
        pignet_gemm128<<<gGemm, T, 0, stream>>>(HHf, WiT + l*DIM*DIM, intWb + l*DIM, H1, nullptr, 1);
        pignet_fill_u32<<<gF2H, T, 0, stream>>>((unsigned*)HA, 0u, (int)ND);
        pignet_scatter<<<gScatC, T, 0, stream>>>(nullptr, nullptr, H1, srcC, dstC, HA, N_INTER);
        pignet_blend<<<gBlend, T, 0, stream>>>(H, HA, intGw + l*2*DIM, intGb + l, 0, N_NODES);
    }

    // -------- i-edge energies --------
    const int* iI = eIe;
    const int* jI = eIe + N_IEDGE;
    pignet_f2h<<<gF2H, T, 0, stream>>>(H, HHf, (int)ND);
    pignet_dist<<<(N_IEDGE + T - 1)/T, T, 0, stream>>>(pos, iI, jI, DA, N_IEDGE);

    const int ntiles = N_IEDGE / 16;                  // 37500
    const int gMlp   = (ntiles + 7) / 8;              // 8 waves/block
    pignet_edge_mlp<<<gMlp, T, 0, stream>>>(HHf, iI, jI, We1T, epsB1, epsW2, epsB2,
                                            nullptr, EX, 0, ntiles);        // eps  -> EX
    pignet_edge_mlp<<<gMlp, T, 0, stream>>>(HHf, iI, jI, Wd1T, dvB1, dvW2, dvB2,
                                            DA, outDvdw, 1, ntiles);        // dvdw -> d_out

    pignet_fill_u32<<<2, T, 0, stream>>>((unsigned*)EG, 0u, N_GRAPHS * 4);
    pignet_energy<<<(N_IEDGE + T - 1)/T, T, 0, stream>>>(DA, vdw, EX, outDvdw, iI, jI, batch,
                                                         mMetal, mHD, mHAc, mHY, hbC, hyC,
                                                         EG, N_IEDGE);
    pignet_final<<<2, T, 0, stream>>>(EG, rotor, roC, outE);
}
